// MultiInputModel_90829968376030
// MI455X (gfx1250) — matrix operand order repaired and checked by reading, not compiled, not run on hardware
//
#include <hip/hip_runtime.h>
#include <hip/hip_bf16.h>
#include <cmath>

// ---------------- problem constants (from reference) ----------------
#define N_NODES   256
#define N_SECRETS 64
#define ENTRIES   27
#define ACTION    4096
#define BATCH     8192
#define STATE_DIM 322

typedef __attribute__((ext_vector_type(16))) _Float16 v16h;
typedef __attribute__((ext_vector_type(8)))  _Float16 v8h;
typedef __attribute__((ext_vector_type(8)))  float    v8f;

union V16U { v16h v; v8h h[2]; };

// ---------------------------------------------------------------
// Kernel 0: convert Wf (f32 [4096,256]) -> f16 in workspace
// ---------------------------------------------------------------
__global__ __launch_bounds__(256) void mim_convert_wf(const float* __restrict__ wf,
                                                      _Float16* __restrict__ wfh,
                                                      int n) {
    int i = blockIdx.x * 256 + threadIdx.x;
    if (i < n) wfh[i] = (_Float16)wf[i];
}

// ---------------------------------------------------------------
// Kernel 1: per-node gather + 27-dot + tanh -> y (f16 [8192,256])
// One block handles 32 batch rows; thread tid owns node n=tid.
// ---------------------------------------------------------------
#define ROWS_PER_BLOCK 32
#define SROW_PAD 324

__global__ __launch_bounds__(256) void mim_node_stage(const float* __restrict__ state,
                                                      const float* __restrict__ W,
                                                      const float* __restrict__ b,
                                                      const int*   __restrict__ idx,
                                                      _Float16*    __restrict__ y) {
    __shared__ float srow[ROWS_PER_BLOCK][SROW_PAD];

    const int tid  = threadIdx.x;
    const int row0 = blockIdx.x * ROWS_PER_BLOCK;

    // Stage 32 state rows into LDS (coalesced global reads).
    const float* gsrc = state + (size_t)row0 * STATE_DIM;
    for (int i = tid; i < ROWS_PER_BLOCK * STATE_DIM; i += 256) {
        int r = i / STATE_DIM;
        int c = i - r * STATE_DIM;
        srow[r][c] = gsrc[i];
    }
    __syncthreads();

    // Thread tid = node n: keep its 27 weights + gather indices in registers.
    const int n = tid;
    float wreg[ENTRIES];
    int   ireg[ENTRIES];
#pragma unroll
    for (int e = 0; e < ENTRIES; ++e) {
        wreg[e] = W[n * ENTRIES + e];
        ireg[e] = idx[n * ENTRIES + e];
    }
    const float bias = b[n];

    for (int r = 0; r < ROWS_PER_BLOCK; ++r) {
        float sum = bias;
#pragma unroll
        for (int e = 0; e < ENTRIES; ++e)
            sum = fmaf(srow[r][ireg[e]], wreg[e], sum);
        y[(size_t)(row0 + r) * N_NODES + n] = (_Float16)tanhf(sum);
    }
}

// ---------------------------------------------------------------
// Kernel 2: out[8192,4096] = 500*sigmoid( y[8192,256] @ WfT )
// Block tile 128(M) x 128(N), 8 waves in a 4(M) x 2(N) grid,
// each wave: 32x64 = 2x4 tiles of v_wmma_f32_16x16x32_f16, K=256.
// y and Wf (f16, 6 MB total) are L2-resident; fragments are loaded
// per-lane straight from global in contiguous k order (lane half h holds K k+16h..k+16h+15).
// ---------------------------------------------------------------
#define TILE_M 128
#define TILE_N 128
#define KDIM   256

__global__ __launch_bounds__(256) void mim_gemm_stage(const _Float16* __restrict__ y,
                                                      const _Float16* __restrict__ wfh,
                                                      float* __restrict__ out) {
    const int lane   = threadIdx.x & 31;
    const int wid    = threadIdx.x >> 5;   // 0..7
    const int waveM  = wid & 3;            // 0..3 -> +32 rows each
    const int waveN  = wid >> 2;           // 0..1 -> +64 cols each
    const int baseM  = blockIdx.y * TILE_M + waveM * 32;
    const int baseN  = blockIdx.x * TILE_N + waveN * 64;

    const int laneLo  = lane & 15;
    const int halfSel = lane >> 4;         // 0: first K-half group, 1: second

    v8f acc[2][4];
#pragma unroll
    for (int mi = 0; mi < 2; ++mi)
#pragma unroll
        for (int ni = 0; ni < 4; ++ni)
            acc[mi][ni] = (v8f){0.f, 0.f, 0.f, 0.f, 0.f, 0.f, 0.f, 0.f};

#pragma unroll
    for (int k = 0; k < KDIM; k += 32) {
        // ---- A fragments: 16x32 f16 tile of y ----
        // lane<16 : row M=laneLo, halves = K k+0..15
        // lane>=16: row M=laneLo, halves = K k+16..31
        v16h afrag[2];
#pragma unroll
        for (int mi = 0; mi < 2; ++mi) {
            const _Float16* pa =
                y + (size_t)(baseM + 16 * mi + laneLo) * KDIM + k + 16 * halfSel;
            V16U a;
            a.h[0] = *(const v8h*)(pa);        // 16B
            a.h[1] = *(const v8h*)(pa + 8);    // 16B
            afrag[mi] = a.v;
        }

        // ---- B fragments: 32x16 f16 tile of Wf^T ----
        // lane<16 : col N=laneLo (Wf row baseN+ni*16+laneLo), halves = K k+0..15
        // lane>=16: same col, halves = K k+16..31
        v16h bfrag[4];
#pragma unroll
        for (int ni = 0; ni < 4; ++ni) {
            const _Float16* pb =
                wfh + (size_t)(baseN + 16 * ni + laneLo) * KDIM + k + 16 * halfSel;
            V16U bvec;
            bvec.h[0] = *(const v8h*)(pb);
            bvec.h[1] = *(const v8h*)(pb + 8);
            bfrag[ni] = bvec.v;
        }

        // ---- 8 WMMAs: D = A x B + C ----
#pragma unroll
        for (int mi = 0; mi < 2; ++mi)
#pragma unroll
            for (int ni = 0; ni < 4; ++ni)
                acc[mi][ni] = __builtin_amdgcn_wmma_f32_16x16x32_f16(
                    /*neg_a=*/false, afrag[mi],
                    /*neg_b=*/false, bfrag[ni],
                    /*c_mod=*/(short)0, acc[mi][ni],
                    /*reuse_a=*/false, /*reuse_b=*/false);
    }

    // ---- Epilogue: 500*sigmoid, per f32 C/D layout:
    // VGPR r of lane: M = r + 8*(lane/16), N = lane%16.
#pragma unroll
    for (int mi = 0; mi < 2; ++mi)
#pragma unroll
        for (int ni = 0; ni < 4; ++ni) {
            const int col = baseN + 16 * ni + laneLo;
#pragma unroll
            for (int r = 0; r < 8; ++r) {
                const int row = baseM + 16 * mi + r + 8 * halfSel;
                float v = acc[mi][ni][r];
                out[(size_t)row * ACTION + col] = 500.f / (1.f + __expf(-v));
            }
        }
}

// ---------------------------------------------------------------
// Host-side launcher. Inputs (setup_inputs order):
//   d_in[0] = state [8192,322] f32
//   d_in[1] = W     [256,27]   f32
//   d_in[2] = b     [256]      f32
//   d_in[3] = Wf    [4096,256] f32
//   d_in[4] = idx   [256,27]   i32
// d_out = [8192,4096] f32
// Workspace layout: y_f16 (4 MB) | wf_f16 (2 MB)
// ---------------------------------------------------------------
extern "C" void kernel_launch(void* const* d_in, const int* in_sizes, int n_in,
                              void* d_out, int out_size, void* d_ws, size_t ws_size,
                              hipStream_t stream) {
    const float* state = (const float*)d_in[0];
    const float* W     = (const float*)d_in[1];
    const float* b     = (const float*)d_in[2];
    const float* Wf    = (const float*)d_in[3];
    const int*   idx   = (const int*)d_in[4];
    float*       out   = (float*)d_out;

    _Float16* y_h  = (_Float16*)d_ws;                                   // 8192*256*2 = 4 MB
    _Float16* wf_h = (_Float16*)((char*)d_ws + (size_t)BATCH * N_NODES * 2); // 2 MB

    // 1) Wf -> f16
    {
        int n = ACTION * N_NODES;
        mim_convert_wf<<<(n + 255) / 256, 256, 0, stream>>>(Wf, wf_h, n);
    }
    // 2) gather + tiny linear + tanh -> y (f16)
    {
        dim3 grid(BATCH / ROWS_PER_BLOCK);
        mim_node_stage<<<grid, 256, 0, stream>>>(state, W, b, idx, y_h);
    }
    // 3) WMMA GEMM + scaled sigmoid -> out
    {
        dim3 grid(ACTION / TILE_N, BATCH / TILE_M);
        mim_gemm_stage<<<grid, 256, 0, stream>>>(y_h, wf_h, out);
    }
}
